// DynamicVit_25744033973208
// MI455X (gfx1250) — matrix operand order repaired and checked by reading, not compiled, not run on hardware
//
#include <hip/hip_runtime.h>
#include <math.h>

#define B_      64
#define NTOK    197
#define NPATCH  196
#define DMODEL  384
#define NHEAD   6
#define HD      64
#define DEPTH_  12
#define NCLS    1000
#define ROWS    (B_*NTOK)    /* 12608 */
#define PROWS   (B_*NPATCH)  /* 12544 */
#define SJ      224          /* padded attention col stride (7*32) */
#define BSLD    40           /* LDS B-tile row stride in halfs (80B: 16 distinct banks) */

typedef __attribute__((ext_vector_type(16))) __bf16        v16bf;
typedef __attribute__((ext_vector_type(8)))  float         v8f;
typedef __attribute__((ext_vector_type(4)))  unsigned int  u32x4;

union V16 { u32x4 q[2]; v16bf v; };

__device__ __forceinline__ unsigned short f2bf(float x){
  unsigned int u = __float_as_uint(x);
  unsigned int r = (u + 0x7FFFu + ((u >> 16) & 1u)) >> 16;
  return (unsigned short)r;
}
__device__ __forceinline__ float bf2f(unsigned short b){
  return __uint_as_float(((unsigned int)b) << 16);
}
__device__ __forceinline__ float gelu_f(float x){
  return 0.5f * x * (1.0f + erff(x * 0.70710678118654752f));
}

// gfx1250 async copy: global -> LDS, 16 bytes per lane, tracked by ASYNCcnt.
__device__ __forceinline__ void async_ld_b128(unsigned lds_off,
                                              const unsigned short* gsrc){
  asm volatile("global_load_async_to_lds_b128 %0, %1, off"
               :: "v"(lds_off), "v"(gsrc) : "memory");
}
__device__ __forceinline__ void wait_async_le1(){
  asm volatile("s_wait_asynccnt 0x1" ::: "memory");
}
__device__ __forceinline__ void wait_async_le0(){
  asm volatile("s_wait_asynccnt 0x0" ::: "memory");
}

// ---------------------------------------------------------------------------
// Batched WMMA GEMM:  C[M,N] = A[M,K] * B[N,K]^T  (bf16 in, f32 acc)
// Block = 8 waves stacked in M: block tile 128(M) x 64(N).
// Each wave: 16x64 (4 accumulators) -> one A fetch feeds 4 WMMAs.
// B tile (64 x 32 halfs) staged in LDS via async copy, double buffered.
// Batch z: A += z*sA, B += z*sB, out offset = (z/bdiv)*sOutO + (z%bdiv)*sInO.
// Requires K % 32 == 0, lda/ldb multiples of 8.
// ---------------------------------------------------------------------------
template<bool BIAS, bool RES, bool GELU_, bool OUTF, bool OUTB>
__global__ __launch_bounds__(256) void gemm_wmma(
    const unsigned short* __restrict__ A, long long sA, int lda,
    const unsigned short* __restrict__ Bm, long long sB, int ldb,
    const float* __restrict__ bias,
    const float* res, int ldr,
    float* outF, int ldf,
    unsigned short* outB, int ldob,
    int bdiv, long long sOutO, long long sInO,
    int M, int N, int K)
{
  __shared__ unsigned short Bs[2][64][BSLD];

  int z = blockIdx.z;
  const unsigned short* Ab = A + (long long)z * sA;
  const unsigned short* Bb = Bm + (long long)z * sB;
  long long offO = (long long)(z / bdiv) * sOutO + (long long)(z % bdiv) * sInO;

  int t = threadIdx.x;
  int wave = t >> 5;
  int lane = t & 31;
  int m0  = blockIdx.y * 128 + wave * 16;
  int n0b = blockIdx.x * 64;

  // B-tile staging addresses for this thread: 256 threads x 16B = 64 rows x 64B
  int srow = t >> 2;            // 0..63  (N-local row)
  int scol = (t & 3) * 8;       // 0,8,16,24 halfs
  int nrow = n0b + srow; if (nrow >= N) nrow = N - 1;
  const unsigned short* bsrc = Bb + (long long)nrow * ldb + scol;
  unsigned lds0 = (unsigned)(unsigned long long)&Bs[0][srow][scol];
  unsigned lds1 = (unsigned)(unsigned long long)&Bs[1][srow][scol];

  // A operand: lane<16 -> M=lane, K interleaved {0-7,16-23}; lane>=16 -> {8-15,24-31}
  int rowA = m0 + (lane & 15); if (rowA >= M) rowA = M - 1;
  int kbA = (lane >> 4) * 8;
  int kbB = (lane >> 4) * 8;    // B operand: K interleaved like A {0-7,16-23}/{8-15,24-31}
  const unsigned short* ap = Ab + (long long)rowA * lda + kbA;
  int blane = lane & 15;

  v8f acc[4];
  for (int j = 0; j < 4; ++j)
    for (int r = 0; r < 8; ++r) acc[j][r] = 0.0f;

  async_ld_b128(lds0, bsrc);          // prime buffer 0 with K-slice 0
  int cur = 0;
  for (int k0 = 0; k0 < K; k0 += 32) {
    int nxt = k0 + 32;
    if (nxt < K) {
      async_ld_b128(cur ? lds0 : lds1, bsrc + nxt);  // prefetch into other buf
      wait_async_le1();                              // oldest (current) landed
    } else {
      wait_async_le0();
    }
    __syncthreads();                  // all waves' slices of Bs[cur] visible

    V16 a;
    a.q[0] = *(const u32x4*)(ap + k0);
    a.q[1] = *(const u32x4*)(ap + k0 + 16);
    const unsigned short* bsp = &Bs[cur][0][0];
#pragma unroll
    for (int j = 0; j < 4; ++j) {
      const unsigned short* bj = bsp + (j * 16 + blane) * BSLD + kbB;
      V16 bvv;
      bvv.q[0] = *(const u32x4*)(bj);
      bvv.q[1] = *(const u32x4*)(bj + 16);
      acc[j] = __builtin_amdgcn_wmma_f32_16x16x32_bf16(false, a.v, false, bvv.v,
                                                       (short)0, acc[j], false, false);
    }
    __syncthreads();                  // reads of Bs[cur] done before overwrite
    cur ^= 1;
  }

  // ---- epilogue (no barriers below; per-lane guards are safe) ------------
  int col0  = n0b + blane;
  int rbase = m0 + ((lane >> 4) << 3);
  for (int j = 0; j < 4; ++j) {
    int col = col0 + j * 16;
    if (col >= N) continue;
    float bvs = BIAS ? bias[col] : 0.0f;
    for (int r = 0; r < 8; ++r) {
      int row = rbase + r;
      if (row < M) {
        float v = acc[j][r] + bvs;
        if (RES)   v += res[(long long)row * ldr + col];
        if (GELU_) v = gelu_f(v);
        if (OUTF)  outF[offO + (long long)row * ldf + col] = v;
        if (OUTB)  outB[offO + (long long)row * ldob + col] = f2bf(v);
      }
    }
  }
}

// ---------------------------------------------------------------------------
// LayerNorm over D=384, block per row, write bf16. mode: 0=direct rows,
// 1=skip-cls (predictor: row r -> (r/196)*197+1+r%196), 2=cls-only (r -> r*197)
// ---------------------------------------------------------------------------
__global__ __launch_bounds__(128) void ln_kernel(
    const float* __restrict__ x, const float* __restrict__ w,
    const float* __restrict__ b, unsigned short* __restrict__ out,
    int M, int mode)
{
  int r = blockIdx.x;
  if (r >= M) return;
  long long ir;
  if (mode == 0)      ir = r;
  else if (mode == 1) ir = (long long)(r / NPATCH) * NTOK + 1 + (r % NPATCH);
  else                ir = (long long)r * NTOK;
  const float* xp = x + ir * DMODEL;
  int tid = threadIdx.x;
  float s = 0.f, ss = 0.f;
  for (int c = tid; c < DMODEL; c += 128) { float v = xp[c]; s += v; ss += v * v; }
  __shared__ float r1[128], r2[128];
  r1[tid] = s; r2[tid] = ss; __syncthreads();
  for (int st = 64; st > 0; st >>= 1) {
    if (tid < st) { r1[tid] += r1[tid + st]; r2[tid] += r2[tid + st]; }
    __syncthreads();
  }
  float mean = r1[0] * (1.0f / DMODEL);
  float var  = r2[0] * (1.0f / DMODEL) - mean * mean;
  float inv  = rsqrtf(var + 1e-5f);
  for (int c = tid; c < DMODEL; c += 128)
    out[(long long)r * DMODEL + c] = f2bf((xp[c] - mean) * inv * w[c] + b[c]);
}

// ---------------------------------------------------------------------------
// Policy softmax: block per (z=b*H+h, i) row of S[z, i, 0..196] (stride SJ).
// Writes bf16 probs, zero-pads cols 197..223.
// ---------------------------------------------------------------------------
__global__ __launch_bounds__(128) void softmax_policy_kernel(
    const float* __restrict__ S, const float* __restrict__ pol,
    unsigned short* __restrict__ P)
{
  int gid = blockIdx.x;
  int i = gid % NTOK;
  int z = gid / NTOK;
  int b = z / NHEAD;
  const float* sp = S + (long long)z * NTOK * SJ + (long long)i * SJ;
  unsigned short* pp = P + (long long)z * NTOK * SJ + (long long)i * SJ;
  int tid = threadIdx.x;
  __shared__ float red[128];
  __shared__ float ebuf[SJ];
  float lm = -3.4e38f;
  for (int j = tid; j < NTOK; j += 128) { float v = sp[j]; lm = v > lm ? v : lm; }
  red[tid] = lm; __syncthreads();
  for (int st = 64; st > 0; st >>= 1) {
    if (tid < st) { float o = red[tid + st]; if (o > red[tid]) red[tid] = o; }
    __syncthreads();
  }
  float m = red[0]; __syncthreads();
  float lsum = 0.f;
  for (int j = tid; j < NTOK; j += 128) {
    float p  = pol[b * NTOK + j];
    float ap = p + (1.0f - p) * ((j == i) ? 1.0f : 0.0f);
    float e  = expf(sp[j] - m) * ap;
    ebuf[j] = e; lsum += e;
  }
  red[tid] = lsum; __syncthreads();
  for (int st = 64; st > 0; st >>= 1) {
    if (tid < st) red[tid] += red[tid + st];
    __syncthreads();
  }
  float inv  = 1.0f / (red[0] + 1e-6f);
  float addc = 1e-6f / (float)NTOK;
  for (int j = tid; j < SJ; j += 128)
    pp[j] = f2bf((j < NTOK) ? (ebuf[j] + addc) * inv : 0.0f);
}

// qkv_bf [B*197, 1152] -> q,k [B,H,197,64] (q scaled by 1/8), vT [B,H,64,224]
__global__ void split_qkv_kernel(const unsigned short* __restrict__ qkv,
    unsigned short* __restrict__ q, unsigned short* __restrict__ k,
    unsigned short* __restrict__ vt)
{
  long long idx = (long long)blockIdx.x * blockDim.x + threadIdx.x;
  long long total = (long long)B_ * NHEAD * HD * SJ;
  if (idx >= total) return;
  int j = (int)(idx % SJ);
  long long rest = idx / SJ;
  int d = (int)(rest % HD); rest /= HD;
  int h = (int)(rest % NHEAD);
  int b = (int)(rest / NHEAD);
  long long zbh = (long long)b * NHEAD + h;
  unsigned short vv = 0;
  if (j < NTOK) vv = qkv[((long long)(b * NTOK + j)) * 1152 + 768 + h * HD + d];
  vt[(zbh * HD + d) * SJ + j] = vv;
  if (j < NTOK) {
    long long qi = (zbh * NTOK + j) * HD + d;
    unsigned short qv = qkv[((long long)(b * NTOK + j)) * 1152 + h * HD + d];
    q[qi] = f2bf(bf2f(qv) * 0.125f);                       // 1/sqrt(64)
    k[qi] = qkv[((long long)(b * NTOK + j)) * 1152 + DMODEL + h * HD + d];
  }
}

// im2col of 224x224x3 into [B*196, 768] bf16 (order c,py,px to match patch_w)
__global__ void im2col_kernel(const float* __restrict__ x,
                              unsigned short* __restrict__ xc)
{
  long long idx = (long long)blockIdx.x * blockDim.x + threadIdx.x;
  if (idx >= (long long)PROWS * 768) return;
  int cc = (int)(idx % 768);
  long long r = idx / 768;
  int t = (int)(r % NPATCH);
  int b = (int)(r / NPATCH);
  int c = cc / 256, py = (cc % 256) / 16, px = cc % 16;
  int row = (t / 14) * 16 + py, col = (t % 14) * 16 + px;
  xc[idx] = f2bf(x[(((long long)b * 3 + c) * 224 + row) * 224 + col]);
}

// h[b,0,:] = cls ; h[b,1+t,:] = patch_out + pos_embed
__global__ void assemble_kernel(const float* __restrict__ po,
    const float* __restrict__ cls, const float* __restrict__ pos,
    float* __restrict__ h)
{
  long long idx = (long long)blockIdx.x * blockDim.x + threadIdx.x;
  if (idx >= (long long)ROWS * DMODEL) return;
  int d = (int)(idx % DMODEL);
  long long r = idx / DMODEL;
  int tok = (int)(r % NTOK);
  int b   = (int)(r / NTOK);
  float v;
  if (tok == 0) v = cls[d];
  else v = po[((long long)b * NPATCH + (tok - 1)) * DMODEL + d]
         + pos[(long long)(tok - 1) * DMODEL + d];
  h[idx] = v;
}

__global__ void init_policy_kernel(float* __restrict__ policy,
                                   float* __restrict__ prev)
{
  int idx = blockIdx.x * blockDim.x + threadIdx.x;
  if (idx < B_ * NTOK)  policy[idx] = 1.0f;
  if (idx < B_ * NPATCH) prev[idx] = 1.0f;
}

// glob[b,c] = sum_t x1[b,t,192+c]*prev[b,t] / sum_t prev[b,t]
__global__ void pool_kernel(const float* __restrict__ x1,
    const float* __restrict__ prev, float* __restrict__ glob)
{
  int idx = blockIdx.x * blockDim.x + threadIdx.x;
  if (idx >= B_ * 192) return;
  int c = idx % 192, b = idx / 192;
  float num = 0.f, den = 0.f;
  for (int t = 0; t < NPATCH; ++t) {
    float p = prev[b * NPATCH + t];
    den += p;
    num += x1[((long long)(b * NPATCH + t)) * DMODEL + 192 + c] * p;
  }
  glob[idx] = num / den;
}

// cat[:, :192] = local, cat[:, 192:] = broadcast glob ; bf16 out
__global__ void concat_kernel(const float* __restrict__ x1,
    const float* __restrict__ glob, unsigned short* __restrict__ cat)
{
  long long idx = (long long)blockIdx.x * blockDim.x + threadIdx.x;
  if (idx >= (long long)PROWS * DMODEL) return;
  int c = (int)(idx % DMODEL);
  long long r = idx / DMODEL;
  int b = (int)(r / NPATCH);
  float v = (c < 192) ? x1[r * DMODEL + c] : glob[b * 192 + (c - 192)];
  cat[idx] = f2bf(v);
}

// final 96->2 matmul + Gumbel argmax (log_softmax shift cancels in argmax);
// keep = hard0 * prev ; update prev and policy in place.
__global__ void score_hard_kernel(const unsigned short* __restrict__ h2,
    const float* __restrict__ w3, const float* __restrict__ b3,
    const float* __restrict__ u, int j,
    float* __restrict__ prev, float* __restrict__ policy)
{
  int idx = blockIdx.x * blockDim.x + threadIdx.x;
  if (idx >= B_ * NPATCH) return;
  int t = idx % NPATCH, b = idx / NPATCH;
  float r0 = b3[0], r1 = b3[1];
  const unsigned short* hp = h2 + (long long)idx * 96;
  for (int k = 0; k < 96; ++k) {
    float v = bf2f(hp[k]);
    r0 += v * w3[k];
    r1 += v * w3[96 + k];
  }
  long long ui = (((long long)j * B_ + b) * NPATCH + t) * 2;
  float g0 = -logf(-logf(u[ui]     + 1e-10f) + 1e-10f);
  float g1 = -logf(-logf(u[ui + 1] + 1e-10f) + 1e-10f);
  float pv = prev[idx];
  float keep = ((r0 + g0) >= (r1 + g1)) ? pv : 0.0f;  // argmax ties -> class 0
  prev[idx] = keep;
  policy[b * NTOK + 1 + t] = keep;
  if (t == 0) policy[b * NTOK] = 1.0f;
}

__global__ void f2bf_kernel(const float* __restrict__ in,
                            unsigned short* __restrict__ outp, long long n)
{
  long long i = (long long)blockIdx.x * blockDim.x + threadIdx.x;
  long long stride = (long long)gridDim.x * blockDim.x;
  for (; i < n; i += stride) outp[i] = f2bf(in[i]);
}

// ---------------------------------------------------------------------------
static inline size_t alignup(size_t x) { return (x + 255) & ~(size_t)255; }
#define GEMM_GRID(M, N, Z) dim3((unsigned)(((N) + 63) / 64), (unsigned)(((M) + 127) / 128), (unsigned)(Z))

extern "C" void kernel_launch(void* const* d_in, const int* in_sizes, int n_in,
                              void* d_out, int out_size, void* d_ws, size_t ws_size,
                              hipStream_t stream)
{
  (void)in_sizes; (void)n_in; (void)out_size;
  const float* x_in  = (const float*)d_in[0];
  const float* u_g   = (const float*)d_in[1];
  const float* patch_w = (const float*)d_in[2];
  const float* patch_b = (const float*)d_in[3];
  const float* cls_t = (const float*)d_in[4];
  const float* pos_e = (const float*)d_in[5];
  const float* ln1_w = (const float*)d_in[6];
  const float* ln1_b = (const float*)d_in[7];
  const float* qkv_w = (const float*)d_in[8];
  const float* proj_w = (const float*)d_in[9];
  const float* proj_b = (const float*)d_in[10];
  const float* ln2_w = (const float*)d_in[11];
  const float* ln2_b = (const float*)d_in[12];
  const float* fc1_w = (const float*)d_in[13];
  const float* fc1_b = (const float*)d_in[14];
  const float* fc2_w = (const float*)d_in[15];
  const float* fc2_b = (const float*)d_in[16];
  const float* pln_w = (const float*)d_in[17];
  const float* pln_b = (const float*)d_in[18];
  const float* pin_w = (const float*)d_in[19];
  const float* pin_b = (const float*)d_in[20];
  const float* pw1 = (const float*)d_in[21];
  const float* pb1 = (const float*)d_in[22];
  const float* pw2 = (const float*)d_in[23];
  const float* pb2 = (const float*)d_in[24];
  const float* pw3 = (const float*)d_in[25];
  const float* pb3 = (const float*)d_in[26];
  const float* norm_w = (const float*)d_in[27];
  const float* norm_b = (const float*)d_in[28];
  const float* head_w = (const float*)d_in[29];
  const float* head_b = (const float*)d_in[30];
  float* out = (float*)d_out;

  char* base = (char*)d_ws;
  size_t off = 0;
  auto carve = [&](size_t bytes) -> void* {
    void* p = base + off; off = alignup(off + bytes); return p;
  };
  typedef unsigned short bf_t;
  // weights (bf16)
  bf_t* patch_wb  = (bf_t*)carve((size_t)DMODEL * 768 * 2);
  bf_t* qkv_wb    = (bf_t*)carve((size_t)DEPTH_ * 1152 * DMODEL * 2);
  bf_t* proj_wb   = (bf_t*)carve((size_t)DEPTH_ * DMODEL * DMODEL * 2);
  bf_t* fc1_wb    = (bf_t*)carve((size_t)DEPTH_ * 1536 * DMODEL * 2);
  bf_t* fc2_wb    = (bf_t*)carve((size_t)DEPTH_ * DMODEL * 1536 * 2);
  bf_t* pin_wb    = (bf_t*)carve((size_t)3 * DMODEL * DMODEL * 2);
  bf_t* p1_wb     = (bf_t*)carve((size_t)3 * 192 * DMODEL * 2);
  bf_t* p2_wb     = (bf_t*)carve((size_t)3 * 96 * 192 * 2);
  bf_t* head_wb   = (bf_t*)carve((size_t)NCLS * DMODEL * 2);
  // activations
  bf_t*  xcol_bf  = (bf_t*)carve((size_t)PROWS * 768 * 2);
  float* patch_o  = (float*)carve((size_t)PROWS * DMODEL * 4);
  float* hbuf     = (float*)carve((size_t)ROWS * DMODEL * 4);
  bf_t*  ln_bf    = (bf_t*)carve((size_t)ROWS * DMODEL * 2);
  bf_t*  qkv_bf   = (bf_t*)carve((size_t)ROWS * 1152 * 2);
  bf_t*  q_bf     = (bf_t*)carve((size_t)B_ * NHEAD * NTOK * HD * 2);
  bf_t*  k_bf     = (bf_t*)carve((size_t)B_ * NHEAD * NTOK * HD * 2);
  bf_t*  vt_bf    = (bf_t*)carve((size_t)B_ * NHEAD * HD * SJ * 2);
  float* Sbuf     = (float*)carve((size_t)B_ * NHEAD * NTOK * SJ * 4);
  bf_t*  P_bf     = (bf_t*)carve((size_t)B_ * NHEAD * NTOK * SJ * 2);
  bf_t*  ao_bf    = (bf_t*)carve((size_t)ROWS * DMODEL * 2);
  bf_t*  mlp_bf   = (bf_t*)carve((size_t)ROWS * 1536 * 2);
  float* x1f      = (float*)carve((size_t)PROWS * DMODEL * 4);
  bf_t*  cat_bf   = (bf_t*)carve((size_t)PROWS * DMODEL * 2);
  bf_t*  h1_bf    = (bf_t*)carve((size_t)PROWS * 192 * 2);
  bf_t*  h2_bf    = (bf_t*)carve((size_t)PROWS * 96 * 2);
  float* globbuf  = (float*)carve((size_t)B_ * 192 * 4);
  float* policy   = (float*)carve((size_t)B_ * NTOK * 4);
  float* prevb    = (float*)carve((size_t)B_ * NPATCH * 4);
  bf_t*  hln_bf   = (bf_t*)carve((size_t)B_ * DMODEL * 2);
  if (off > ws_size) return;  // workspace too small: cannot run

  // ---- convert weights to bf16 -------------------------------------------
  auto conv = [&](const float* s, bf_t* d, long long n) {
    f2bf_kernel<<<2048, 256, 0, stream>>>(s, d, n);
  };
  conv(patch_w, patch_wb, (long long)DMODEL * 768);
  conv(qkv_w,  qkv_wb,  (long long)DEPTH_ * 1152 * DMODEL);
  conv(proj_w, proj_wb, (long long)DEPTH_ * DMODEL * DMODEL);
  conv(fc1_w,  fc1_wb,  (long long)DEPTH_ * 1536 * DMODEL);
  conv(fc2_w,  fc2_wb,  (long long)DEPTH_ * DMODEL * 1536);
  conv(pin_w,  pin_wb,  (long long)3 * DMODEL * DMODEL);
  conv(pw1,    p1_wb,   (long long)3 * 192 * DMODEL);
  conv(pw2,    p2_wb,   (long long)3 * 96 * 192);
  conv(head_w, head_wb, (long long)NCLS * DMODEL);

  // ---- patch embedding ----------------------------------------------------
  {
    long long n = (long long)PROWS * 768;
    im2col_kernel<<<(unsigned)((n + 255) / 256), 256, 0, stream>>>(x_in, xcol_bf);
    gemm_wmma<true, false, false, true, false><<<GEMM_GRID(PROWS, DMODEL, 1), 256, 0, stream>>>(
        xcol_bf, 0, 768, patch_wb, 0, 768, patch_b,
        nullptr, 0, patch_o, DMODEL, nullptr, 0,
        1, 0, 0, PROWS, DMODEL, 768);
    long long m = (long long)ROWS * DMODEL;
    assemble_kernel<<<(unsigned)((m + 255) / 256), 256, 0, stream>>>(patch_o, cls_t, pos_e, hbuf);
    init_policy_kernel<<<(B_ * NTOK + 255) / 256, 256, 0, stream>>>(policy, prevb);
  }

  int pcount = 0;
  for (int i = 0; i < DEPTH_; ++i) {
    if (i == 3 || i == 6 || i == 9) {
      int j = pcount++;
      // predictor
      ln_kernel<<<PROWS, 128, 0, stream>>>(hbuf, pln_w + j * DMODEL, pln_b + j * DMODEL,
                                           ln_bf, PROWS, 1);
      gemm_wmma<true, false, true, true, false><<<GEMM_GRID(PROWS, DMODEL, 1), 256, 0, stream>>>(
          ln_bf, 0, DMODEL, pin_wb + (long long)j * DMODEL * DMODEL, 0, DMODEL,
          pin_b + j * DMODEL, nullptr, 0, x1f, DMODEL, nullptr, 0,
          1, 0, 0, PROWS, DMODEL, DMODEL);
      pool_kernel<<<(B_ * 192 + 255) / 256, 256, 0, stream>>>(x1f, prevb, globbuf);
      {
        long long n = (long long)PROWS * DMODEL;
        concat_kernel<<<(unsigned)((n + 255) / 256), 256, 0, stream>>>(x1f, globbuf, cat_bf);
      }
      gemm_wmma<true, false, true, false, true><<<GEMM_GRID(PROWS, 192, 1), 256, 0, stream>>>(
          cat_bf, 0, DMODEL, p1_wb + (long long)j * 192 * DMODEL, 0, DMODEL,
          pb1 + j * 192, nullptr, 0, nullptr, 0, h1_bf, 192,
          1, 0, 0, PROWS, 192, DMODEL);
      gemm_wmma<true, false, true, false, true><<<GEMM_GRID(PROWS, 96, 1), 256, 0, stream>>>(
          h1_bf, 0, 192, p2_wb + (long long)j * 96 * 192, 0, 192,
          pb2 + j * 96, nullptr, 0, nullptr, 0, h2_bf, 96,
          1, 0, 0, PROWS, 96, 192);
      score_hard_kernel<<<(B_ * NPATCH + 255) / 256, 256, 0, stream>>>(
          h2_bf, pw3 + j * 192, pb3 + j * 2, u_g, j, prevb, policy);
    }

    // ---- attention --------------------------------------------------------
    ln_kernel<<<ROWS, 128, 0, stream>>>(hbuf, ln1_w + i * DMODEL, ln1_b + i * DMODEL,
                                        ln_bf, ROWS, 0);
    gemm_wmma<false, false, false, false, true><<<GEMM_GRID(ROWS, 1152, 1), 256, 0, stream>>>(
        ln_bf, 0, DMODEL, qkv_wb + (long long)i * 1152 * DMODEL, 0, DMODEL,
        nullptr, nullptr, 0, nullptr, 0, qkv_bf, 1152,
        1, 0, 0, ROWS, 1152, DMODEL);
    {
      long long n = (long long)B_ * NHEAD * HD * SJ;
      split_qkv_kernel<<<(unsigned)((n + 255) / 256), 256, 0, stream>>>(qkv_bf, q_bf, k_bf, vt_bf);
    }
    gemm_wmma<false, false, false, true, false><<<GEMM_GRID(NTOK, NTOK, B_ * NHEAD), 256, 0, stream>>>(
        q_bf, (long long)NTOK * HD, HD, k_bf, (long long)NTOK * HD, HD,
        nullptr, nullptr, 0, Sbuf, SJ, nullptr, 0,
        1, (long long)NTOK * SJ, 0, NTOK, NTOK, HD);
    softmax_policy_kernel<<<B_ * NHEAD * NTOK, 128, 0, stream>>>(Sbuf, policy, P_bf);
    gemm_wmma<false, false, false, false, true><<<GEMM_GRID(NTOK, HD, B_ * NHEAD), 256, 0, stream>>>(
        P_bf, (long long)NTOK * SJ, SJ, vt_bf, (long long)HD * SJ, SJ,
        nullptr, nullptr, 0, nullptr, 0, ao_bf, DMODEL,
        NHEAD, (long long)NTOK * DMODEL, HD, NTOK, HD, SJ);
    gemm_wmma<true, true, false, true, false><<<GEMM_GRID(ROWS, DMODEL, 1), 256, 0, stream>>>(
        ao_bf, 0, DMODEL, proj_wb + (long long)i * DMODEL * DMODEL, 0, DMODEL,
        proj_b + i * DMODEL, hbuf, DMODEL, hbuf, DMODEL, nullptr, 0,
        1, 0, 0, ROWS, DMODEL, DMODEL);

    // ---- MLP --------------------------------------------------------------
    ln_kernel<<<ROWS, 128, 0, stream>>>(hbuf, ln2_w + i * DMODEL, ln2_b + i * DMODEL,
                                        ln_bf, ROWS, 0);
    gemm_wmma<true, false, true, false, true><<<GEMM_GRID(ROWS, 1536, 1), 256, 0, stream>>>(
        ln_bf, 0, DMODEL, fc1_wb + (long long)i * 1536 * DMODEL, 0, DMODEL,
        fc1_b + i * 1536, nullptr, 0, nullptr, 0, mlp_bf, 1536,
        1, 0, 0, ROWS, 1536, DMODEL);
    gemm_wmma<true, true, false, true, false><<<GEMM_GRID(ROWS, DMODEL, 1), 256, 0, stream>>>(
        mlp_bf, 0, 1536, fc2_wb + (long long)i * DMODEL * 1536, 0, 1536,
        fc2_b + i * DMODEL, hbuf, DMODEL, hbuf, DMODEL, nullptr, 0,
        1, 0, 0, ROWS, DMODEL, 1536);
  }

  // ---- final LN (cls rows only) + head -----------------------------------
  ln_kernel<<<B_, 128, 0, stream>>>(hbuf, norm_w, norm_b, hln_bf, B_, 2);
  gemm_wmma<true, false, false, true, false><<<GEMM_GRID(B_, NCLS, 1), 256, 0, stream>>>(
      hln_bf, 0, DMODEL, head_wb, 0, DMODEL, head_b,
      nullptr, 0, out, NCLS, nullptr, 0,
      1, 0, 0, B_, NCLS, DMODEL);
}
